// AttentionEncoderLayer_24077586661655
// MI455X (gfx1250) — matrix operand order repaired and checked by reading, not compiled, not run on hardware
//
#include <hip/hip_runtime.h>

// ---------------------------------------------------------------------------
// AttentionEncoderLayer for MI455X (gfx1250, wave32, WMMA).
// N = F*B = 8192 tokens, H = 512, NH = 8, dh = 64, C = 8 kv rows per token.
// All reshapes in the reference are reinterpretations, so flat row order of
// query/key/value/output coincide; no permutes needed.
// GEMMs: v_wmma_f32_16x16x32_bf16, double-buffered LDS; weight tiles staged
// with GLOBAL_LOAD_ASYNC_TO_LDS_B128 (ASYNCcnt), activations converted
// f32->bf16 in flight through VGPRs.
// ---------------------------------------------------------------------------

typedef __attribute__((ext_vector_type(16))) __bf16 v16bf;
typedef __attribute__((ext_vector_type(8)))  float  v8f;

#define BM 128
#define BN 128
#define BK 32
#define SA 40   // padded LDS row stride in bf16 elems (80 bytes, 16B-aligned rows)

// ---- stage a 128x32 f32 tile -> bf16 LDS (convert on the fly) -------------
__device__ __forceinline__ void stageA(const float* __restrict__ A, int K,
                                       int m0, int k0, __bf16* s, int t) {
    int r = t >> 1;
    int c = (t & 1) * 16;                   // 16 floats per thread
    const float4* src = reinterpret_cast<const float4*>(A + (size_t)(m0 + r) * K + k0 + c);
    __bf16* dst = s + r * SA + c;
#pragma unroll
    for (int j = 0; j < 4; ++j) {
        float4 v = src[j];
        dst[4 * j + 0] = (__bf16)v.x;
        dst[4 * j + 1] = (__bf16)v.y;
        dst[4 * j + 2] = (__bf16)v.z;
        dst[4 * j + 3] = (__bf16)v.w;
    }
}

// ---- stage a 128x32 bf16 weight tile -> LDS via async DMA (ASYNCcnt) ------
// ISA: the instruction offset is added to both the LDS and the memory address, so one
// address setup serves both 16B chunks via the instruction offset.
__device__ __forceinline__ void stageB_async(const __bf16* __restrict__ W, int K,
                                             int n0, int k0, __bf16* s, int t) {
    int r = t >> 1;
    int c = (t & 1) * 16;                   // 16 bf16 (32B) per thread
    const __bf16* src = W + (size_t)(n0 + r) * K + k0 + c;
    unsigned lds = (unsigned)(size_t)(s + r * SA + c);  // low 32b = LDS offset
    asm volatile("global_load_async_to_lds_b128 %0, %1, off"
                 :: "v"(lds), "v"(src) : "memory");
    asm volatile("global_load_async_to_lds_b128 %0, %1, off offset:16"
                 :: "v"(lds), "v"(src) : "memory");
}

__device__ __forceinline__ void wait_async0() {
    asm volatile("s_wait_asynccnt 0x0" ::: "memory");
}

// A fragment (16x32 bf16): lane = M (mod 16); lo lanes K{0..7,16..23},
// hi lanes K{8..15,24..31}  (ISA 7.12.2 "16-bit A-Matrix 16x32").
__device__ __forceinline__ v16bf loadAfrag(const __bf16* s, int row, int lhi) {
    const uint4* p = reinterpret_cast<const uint4*>(s + row * SA + lhi * 8);
    v16bf a;
    reinterpret_cast<uint4*>(&a)[0] = p[0];   // K 0..7 / 8..15
    reinterpret_cast<uint4*>(&a)[1] = p[2];   // +32B: K 16..23 / 24..31
    return a;
}

// B fragment (32x16 bf16): lane = N (mod 16); lo lanes K{0..7,16..23},
// hi lanes K{8..15,24..31} (same k order as the A fragment).
__device__ __forceinline__ v16bf loadBfrag(const __bf16* s, int row, int lhi) {
    const uint4* p = reinterpret_cast<const uint4*>(s + row * SA + lhi * 8);
    v16bf b;
    reinterpret_cast<uint4*>(&b)[0] = p[0];
    reinterpret_cast<uint4*>(&b)[1] = p[2];
    return b;
}

// ---------------------------------------------------------------------------
// GEMM: out[M,N] = A[M,K] (f32) * W[N,K]^T (bf16) + bias, optional ReLU,
// output either f32 or bf16. WMMA f32<-bf16, double-buffered LDS.
// Block 256 = 8 waves, block tile 128x128, wave tile 32x64 (2x4 WMMA accs).
// ---------------------------------------------------------------------------
template <int RELU, int OUTBF>
__global__ __launch_bounds__(256) void gemm_bias_wmma(
    const float* __restrict__ A, const __bf16* __restrict__ W,
    const float* __restrict__ bias, void* __restrict__ out,
    int M, int N, int K) {
    __shared__ __bf16 sA[2][BM * SA];
    __shared__ __bf16 sB[2][BN * SA];

    const int t    = threadIdx.x;
    const int n0   = blockIdx.x * BN;
    const int m0   = blockIdx.y * BM;
    const int wid  = t >> 5;
    const int lane = t & 31;
    const int wm   = wid & 3;    // 4 waves along M -> 32 rows each
    const int wn   = wid >> 2;   // 2 waves along N -> 64 cols each
    const int lrow = lane & 15;
    const int lhi  = lane >> 4;

    const v8f vzero = {0.f, 0.f, 0.f, 0.f, 0.f, 0.f, 0.f, 0.f};
    v8f acc[2][4];
#pragma unroll
    for (int i = 0; i < 2; ++i)
#pragma unroll
        for (int j = 0; j < 4; ++j) acc[i][j] = vzero;

    const int KT = K / BK;
    stageA(A, K, m0, 0, sA[0], t);
    stageB_async(W, K, n0, 0, sB[0], t);
    wait_async0();
    __syncthreads();

    for (int kt = 0; kt < KT; ++kt) {
        const int cur = kt & 1;
        if (kt + 1 < KT) {
            stageA(A, K, m0, (kt + 1) * BK, sA[cur ^ 1], t);
            stageB_async(W, K, n0, (kt + 1) * BK, sB[cur ^ 1], t);
        }
        v16bf af[2], bfr[4];
#pragma unroll
        for (int mi = 0; mi < 2; ++mi)
            af[mi] = loadAfrag(sA[cur], wm * 32 + mi * 16 + lrow, lhi);
#pragma unroll
        for (int ni = 0; ni < 4; ++ni)
            bfr[ni] = loadBfrag(sB[cur], wn * 64 + ni * 16 + lrow, lhi);
#pragma unroll
        for (int mi = 0; mi < 2; ++mi)
#pragma unroll
            for (int ni = 0; ni < 4; ++ni)
                acc[mi][ni] = __builtin_amdgcn_wmma_f32_16x16x32_bf16(
                    false, af[mi], false, bfr[ni], (short)0, acc[mi][ni],
                    false, false);
        wait_async0();           // async weight copy overlapped with the WMMAs
        __syncthreads();
    }

    // Epilogue. C/D layout: lane -> N (mod 16), VGPR j -> M = j + 8*lhi.
#pragma unroll
    for (int mi = 0; mi < 2; ++mi)
#pragma unroll
        for (int ni = 0; ni < 4; ++ni) {
            const int col = n0 + wn * 64 + ni * 16 + lrow;
            const float bv = bias[col];
#pragma unroll
            for (int j = 0; j < 8; ++j) {
                const int row = m0 + wm * 32 + mi * 16 + lhi * 8 + j;
                float v = acc[mi][ni][j] + bv;
                if (RELU) v = fmaxf(v, 0.f);
                const size_t idx = (size_t)row * N + col;
                if (OUTBF)
                    ((__bf16*)out)[idx] = (__bf16)v;
                else
                    ((float*)out)[idx] = v;
            }
        }
}

// ---------------------------------------------------------------------------
// Tiny kernels
// ---------------------------------------------------------------------------
__global__ void f32_to_bf16(const float* __restrict__ in,
                            __bf16* __restrict__ out, int n) {
    int i = blockIdx.x * 256 + threadIdx.x;
    if (i < n) out[i] = (__bf16)in[i];
}

// One wave32 per (token, head): softmax over C=8 keys, dh=64 (2 dims/lane).
__global__ __launch_bounds__(256) void attn_small(
    const float* __restrict__ Qf, const __bf16* __restrict__ Kb,
    const __bf16* __restrict__ Vb, float* __restrict__ ctx) {
    const int wid  = threadIdx.x >> 5;
    const int lane = threadIdx.x & 31;
    const int w = blockIdx.x * 8 + wid;
    const int n = w >> 3;
    const int h = w & 7;
    const size_t qoff = (size_t)n * 512 + h * 64 + lane * 2;
    const float q0 = Qf[qoff], q1 = Qf[qoff + 1];

    float sc[8];
#pragma unroll
    for (int c = 0; c < 8; ++c) {
        const size_t koff = ((size_t)n * 8 + c) * 512 + h * 64 + lane * 2;
        float s = q0 * (float)Kb[koff] + q1 * (float)Kb[koff + 1];
#pragma unroll
        for (int off = 16; off > 0; off >>= 1) s += __shfl_xor(s, off, 32);
        sc[c] = s * 0.125f;                 // 1/sqrt(dh), dh=64
    }
    float m = sc[0];
#pragma unroll
    for (int c = 1; c < 8; ++c) m = fmaxf(m, sc[c]);
    float sum = 0.f;
#pragma unroll
    for (int c = 0; c < 8; ++c) { sc[c] = __expf(sc[c] - m); sum += sc[c]; }
    const float inv = 1.0f / sum;
    float o0 = 0.f, o1 = 0.f;
#pragma unroll
    for (int c = 0; c < 8; ++c) {
        const size_t voff = ((size_t)n * 8 + c) * 512 + h * 64 + lane * 2;
        const float p = sc[c] * inv;
        o0 += p * (float)Vb[voff];
        o1 += p * (float)Vb[voff + 1];
    }
    ctx[qoff] = o0;
    ctx[qoff + 1] = o1;
}

// out = LayerNorm(a + b) * g + beta, one row (H=512) per 256-thread block.
__global__ __launch_bounds__(256) void ln_residual(
    const float* __restrict__ a, const float* __restrict__ b,
    const float* __restrict__ g, const float* __restrict__ beta,
    float* __restrict__ out) {
    const int t = threadIdx.x;
    const size_t base = (size_t)blockIdx.x * 512;
    const float x0 = a[base + t] + b[base + t];
    const float x1 = a[base + 256 + t] + b[base + 256 + t];
    float s = x0 + x1;
    float q = x0 * x0 + x1 * x1;
#pragma unroll
    for (int off = 16; off > 0; off >>= 1) {
        s += __shfl_xor(s, off, 32);
        q += __shfl_xor(q, off, 32);
    }
    __shared__ float ss[8], sq[8];
    const int wid = t >> 5, lane = t & 31;
    if (lane == 0) { ss[wid] = s; sq[wid] = q; }
    __syncthreads();
    float S = 0.f, Q = 0.f;
#pragma unroll
    for (int i = 0; i < 8; ++i) { S += ss[i]; Q += sq[i]; }
    const float mean = S * (1.0f / 512.0f);
    const float var  = Q * (1.0f / 512.0f) - mean * mean;
    const float r = rsqrtf(var + 1e-5f);
    out[base + t]       = (x0 - mean) * r * g[t]       + beta[t];
    out[base + 256 + t] = (x1 - mean) * r * g[256 + t] + beta[256 + t];
}

// ---------------------------------------------------------------------------
extern "C" void kernel_launch(void* const* d_in, const int* in_sizes, int n_in,
                              void* d_out, int out_size, void* d_ws,
                              size_t ws_size, hipStream_t stream) {
    (void)in_sizes; (void)n_in; (void)out_size; (void)ws_size;

    const float* query = (const float*)d_in[0];   // [8192, 512] flat
    const float* key   = (const float*)d_in[1];   // [65536, 512] flat
    const float* value = (const float*)d_in[2];
    const float* Wq = (const float*)d_in[3];  const float* bq = (const float*)d_in[4];
    const float* Wk = (const float*)d_in[5];  const float* bk = (const float*)d_in[6];
    const float* Wv = (const float*)d_in[7];  const float* bv = (const float*)d_in[8];
    const float* Wo = (const float*)d_in[9];  const float* bo = (const float*)d_in[10];
    const float* g1 = (const float*)d_in[11]; const float* be1 = (const float*)d_in[12];
    const float* W1 = (const float*)d_in[13]; const float* b1 = (const float*)d_in[14];
    const float* W2 = (const float*)d_in[15]; const float* b2 = (const float*)d_in[16];
    const float* g2 = (const float*)d_in[17]; const float* be2 = (const float*)d_in[18];

    const int Ntok = 8192;    // F*B
    const int Nkv  = 65536;   // F*B*C
    const int H = 512, Hf = 1024;

    char* ws = (char*)d_ws;
    // Phase-1 buffers
    __bf16* Kbf = (__bf16*)(ws + (size_t)0);           // 64 MB
    __bf16* Vbf = (__bf16*)(ws + (size_t)67108864);    // 64 MB
    float*  Qf  = (float*) (ws + (size_t)134217728);   // 16 MB
    float*  ctx = (float*) (ws + (size_t)150994944);   // 16 MB
    __bf16* wqb = (__bf16*)(ws + (size_t)167772160);   // bf16 weights (4 MB)
    __bf16* wkb = wqb + 262144;
    __bf16* wvb = wkb + 262144;
    __bf16* wob = wvb + 262144;
    __bf16* w1b = wob + 262144;
    __bf16* w2b = w1b + 524288;
    // Phase-2 buffers reuse the (then-dead) Kbf/Vbf regions
    float* x   = (float*)(ws + (size_t)0);             // Wo output   (16 MB)
    float* ff2 = (float*)(ws + (size_t)16777216);      // FFN output  (16 MB)
    float* ff1 = (float*)(ws + (size_t)33554432);      // ReLU(h1 W1) (32 MB)
    float* h1  = (float*)(ws + (size_t)67108864);      // LN1 output  (16 MB)

    const dim3 blk(256);

    // Pre-convert weights to bf16 (tiny, once per call)
    f32_to_bf16<<<(262144 + 255) / 256, blk, 0, stream>>>(Wq, wqb, 262144);
    f32_to_bf16<<<(262144 + 255) / 256, blk, 0, stream>>>(Wk, wkb, 262144);
    f32_to_bf16<<<(262144 + 255) / 256, blk, 0, stream>>>(Wv, wvb, 262144);
    f32_to_bf16<<<(262144 + 255) / 256, blk, 0, stream>>>(Wo, wob, 262144);
    f32_to_bf16<<<(524288 + 255) / 256, blk, 0, stream>>>(W1, w1b, 524288);
    f32_to_bf16<<<(524288 + 255) / 256, blk, 0, stream>>>(W2, w2b, 524288);

    // K/V projections (the big GEMMs): output bf16 to halve traffic/workspace
    gemm_bias_wmma<0, 1><<<dim3(H / BN, Nkv / BM), blk, 0, stream>>>(
        key, wkb, bk, Kbf, Nkv, H, H);
    gemm_bias_wmma<0, 1><<<dim3(H / BN, Nkv / BM), blk, 0, stream>>>(
        value, wvb, bv, Vbf, Nkv, H, H);
    // Q projection (f32 out)
    gemm_bias_wmma<0, 0><<<dim3(H / BN, Ntok / BM), blk, 0, stream>>>(
        query, wqb, bq, Qf, Ntok, H, H);

    // Per-token attention over C=8 cached rows
    attn_small<<<dim3(Ntok), blk, 0, stream>>>(Qf, Kbf, Vbf, ctx);

    // Output projection, residual + LN1
    gemm_bias_wmma<0, 0><<<dim3(H / BN, Ntok / BM), blk, 0, stream>>>(
        ctx, wob, bo, x, Ntok, H, H);
    ln_residual<<<dim3(Ntok), blk, 0, stream>>>(query, x, g1, be1, h1);

    // FFN: ReLU(h1 W1^T + b1) W2^T + b2, residual + LN2 -> out
    gemm_bias_wmma<1, 0><<<dim3(Hf / BN, Ntok / BM), blk, 0, stream>>>(
        h1, w1b, b1, ff1, Ntok, Hf, H);
    gemm_bias_wmma<0, 0><<<dim3(H / BN, Ntok / BM), blk, 0, stream>>>(
        ff1, w2b, b2, ff2, Ntok, H, Hf);
    ln_residual<<<dim3(Ntok), blk, 0, stream>>>(h1, ff2, g2, be2, (float*)d_out);
}
